// FullyConnectedTree_63127429317268
// MI455X (gfx1250) — matrix operand order repaired and checked by reading, not compiled, not run on hardware
//
#include <hip/hip_runtime.h>

typedef __attribute__((ext_vector_type(2)))  float    v2f;
typedef __attribute__((ext_vector_type(4)))  float    v4f;
typedef __attribute__((ext_vector_type(8)))  float    v8f;
typedef __attribute__((ext_vector_type(16))) _Float16 v16h;

__device__ __forceinline__ float sigmoid_f(float z) {
  return 1.0f / (1.0f + expf(-z));
}

// ---------------------------------------------------------------------------
// Prep: fold the 3 affine layers into a single matvec  out = x . v + c
//   W_l = sigmoid(el_l/3) * es_l ;  a_l = sigmoid(b_l)*3 - 1
//   v = W0 @ W1 @ w2 ;  c = a0@W1@w2 + a1@w2 + a2
// One block of 64 threads; negligible cost.
// ---------------------------------------------------------------------------
__global__ __launch_bounds__(64) void fct_prep(
    const float* __restrict__ el0, const float* __restrict__ el1,
    const float* __restrict__ el2, const float* __restrict__ es0,
    const float* __restrict__ es1, const float* __restrict__ es2,
    const float* __restrict__ b0,  const float* __restrict__ b1,
    const float* __restrict__ b2,  float* __restrict__ vc) {
  __shared__ float w2s[64];
  __shared__ float us[64];
  __shared__ float ps[64];
  const int t = threadIdx.x;           // 0..63
  const float inv_temp = 1.0f / 3.0f;

  float w2 = sigmoid_f(el2[t] * inv_temp) * es2[t];   // el2 is (64,1)
  w2s[t] = w2;
  __syncthreads();

  float u = 0.0f;                      // u = W1 @ w2
  for (int j = 0; j < 64; ++j)
    u = fmaf(sigmoid_f(el1[t * 64 + j] * inv_temp) * es1[t * 64 + j], w2s[j], u);
  us[t] = u;
  __syncthreads();

  float v = 0.0f;                      // v = W0 @ u
  for (int j = 0; j < 64; ++j)
    v = fmaf(sigmoid_f(el0[t * 64 + j] * inv_temp) * es0[t * 64 + j], us[j], v);
  vc[t] = v;

  float a0 = sigmoid_f(b0[t]) * 3.0f - 1.0f;
  float a1 = sigmoid_f(b1[t]) * 3.0f - 1.0f;
  ps[t] = a0 * us[t] + a1 * w2s[t];
  __syncthreads();
  if (t == 0) {
    float c = sigmoid_f(b2[0]) * 3.0f - 1.0f;
    for (int j = 0; j < 64; ++j) c += ps[j];
    vc[64] = c;
  }
}

// ---------------------------------------------------------------------------
// Main: streaming matvec via WMMA. One wave computes a 16-row tile:
//   D(16x16) += A(16x4 of x) * B(4x16 = v broadcast to all columns)
// 16 K-steps over 4 accumulator chains cover K=64; every column of D equals x_tile . v.
// ---------------------------------------------------------------------------
__global__ __launch_bounds__(256) void fct_matvec(
    const float* __restrict__ x, const float* __restrict__ vc,
    float* __restrict__ out, int ntiles) {
  const int lane   = threadIdx.x & 31;
  const int wave   = threadIdx.x >> 5;
  const int h      = lane >> 4;        // K-half selector (0 or 1)
  const int mlane  = lane & 15;        // A-matrix row within tile
  const int gwave  = blockIdx.x * (blockDim.x >> 5) + wave;
  const int nwaves = gridDim.x * (blockDim.x >> 5);

#if __has_builtin(__builtin_amdgcn_wmma_f32_16x16x4_f32)
  // --- fp32 WMMA path (exact precision) ---
  // B 4x16 layout: VGPR0/1 hold K = 2h + {0,1} for column N = lane&15.
  // Replicate v across all columns -> b is independent of N.
  v2f b[16];
  #pragma unroll
  for (int k4 = 0; k4 < 16; ++k4)
    b[k4] = *(const v2f*)(vc + 4 * k4 + 2 * h);
  const float c = vc[64];

  for (int tile = gwave; tile < ntiles; tile += nwaves) {
    // A 16x4 layout: lane holds row (lane&15), K = 2h + {0,1}
    const float* xrow = x + (size_t)(tile * 16 + mlane) * 64 + 2 * h;
    v2f a[16];
    #pragma unroll
    for (int k4 = 0; k4 < 16; ++k4)
      a[k4] = __builtin_nontemporal_load((const v2f*)(xrow + 4 * k4));

    v8f accs[4];
    accs[0] = (v8f){c, c, c, c, c, c, c, c};   // bias folded into C
    accs[1] = (v8f){};
    accs[2] = (v8f){};
    accs[3] = (v8f){};
    #pragma unroll
    for (int k4 = 0; k4 < 16; ++k4)
      accs[k4 & 3] = __builtin_amdgcn_wmma_f32_16x16x4_f32(
          false, a[k4], false, b[k4], (short)0, accs[k4 & 3], false, false);
    v8f d = (accs[0] + accs[1]) + (accs[2] + accs[3]);

    // D layout: lane 0 holds rows 0..7 (N=0), lane 16 holds rows 8..15.
    if (mlane == 0) {
      float* o = out + tile * 16 + 8 * h;
      v4f lo = {d[0], d[1], d[2], d[3]};
      v4f hi = {d[4], d[5], d[6], d[7]};
      __builtin_nontemporal_store(lo, (v4f*)o);
      __builtin_nontemporal_store(hi, (v4f*)(o + 4));
    }
  }
#else
  // --- f16 WMMA fallback (confirmed builtin) ---
  // B 32x16 layout: per lane, elements 0..7 = K = 8h + {0..7}, elements 8..15 = K = 16 + 8h + {0..7}.
  v16h bf[2];
  #pragma unroll
  for (int kb = 0; kb < 2; ++kb) {
    #pragma unroll
    for (int i = 0; i < 16; ++i)
      bf[kb][i] = (_Float16)vc[32 * kb + 8 * h + i + (i >= 8 ? 8 : 0)];
  }
  const float c = vc[64];

  for (int tile = gwave; tile < ntiles; tile += nwaves) {
    const float* xrow = x + (size_t)(tile * 16 + mlane) * 64;
    v8f acc = (v8f){c, c, c, c, c, c, c, c};
    #pragma unroll
    for (int kb = 0; kb < 2; ++kb) {
      // A 16x32 layout: elements 0..7 = K = 32kb+8h+{0..7},
      //                 elements 8..15 = K = 32kb+16+8h+{0..7}
      const float* p0 = xrow + 32 * kb + 8 * h;
      v4f f0 = __builtin_nontemporal_load((const v4f*)p0);
      v4f f1 = __builtin_nontemporal_load((const v4f*)(p0 + 4));
      v4f f2 = __builtin_nontemporal_load((const v4f*)(p0 + 16));
      v4f f3 = __builtin_nontemporal_load((const v4f*)(p0 + 20));
      v16h af;
      #pragma unroll
      for (int i = 0; i < 4; ++i) {
        af[i]      = (_Float16)f0[i];
        af[i + 4]  = (_Float16)f1[i];
        af[i + 8]  = (_Float16)f2[i];
        af[i + 12] = (_Float16)f3[i];
      }
      acc = __builtin_amdgcn_wmma_f32_16x16x32_f16(false, af, false, bf[kb],
                                                   (short)0, acc, false, false);
    }
    if (mlane == 0) {
      float* o = out + tile * 16 + 8 * h;
      v4f lo = {acc[0], acc[1], acc[2], acc[3]};
      v4f hi = {acc[4], acc[5], acc[6], acc[7]};
      __builtin_nontemporal_store(lo, (v4f*)o);
      __builtin_nontemporal_store(hi, (v4f*)(o + 4));
    }
  }
#endif
}

extern "C" void kernel_launch(void* const* d_in, const int* in_sizes, int n_in,
                              void* d_out, int out_size, void* d_ws, size_t ws_size,
                              hipStream_t stream) {
  const float* x   = (const float*)d_in[0];
  const float* el0 = (const float*)d_in[1];
  const float* el1 = (const float*)d_in[2];
  const float* el2 = (const float*)d_in[3];
  const float* es0 = (const float*)d_in[4];
  const float* es1 = (const float*)d_in[5];
  const float* es2 = (const float*)d_in[6];
  const float* b0  = (const float*)d_in[7];
  const float* b1  = (const float*)d_in[8];
  const float* b2  = (const float*)d_in[9];
  float* out = (float*)d_out;
  float* vc  = (float*)d_ws;          // 64 floats v + 1 float c

  const int nrows  = in_sizes[0] / 64;  // 1,048,576
  const int ntiles = nrows >> 4;        // 65,536 tiles of 16 rows

  fct_prep<<<1, 64, 0, stream>>>(el0, el1, el2, es0, es1, es2, b0, b1, b2, vc);

  int blocks = (ntiles + 7) / 8;
  if (blocks > 4096) blocks = 4096;     // 32768 waves, 2 tiles each
  fct_matvec<<<blocks, 256, 0, stream>>>(x, vc, out, ntiles);
}
